// GwACAttention_28123445854575
// MI455X (gfx1250) — compile-verified, matrix operand order checked by reading, not run on hardware
//
#include <hip/hip_runtime.h>
#include <hip/hip_bf16.h>
#include <math.h>

// ---------------- problem constants (from reference setup_inputs) -------------
#define NN    512
#define DEGC  8
#define FF    128
#define HH    256
#define MMSZ  256
#define OO    64
#define PPSZ  2
#define SSTART 64
#define HISTN 10
#define STEPS (10 * NN)        // 5120 steps; queue never empties (pop1/push8)
#define MSG_SAVE 640           // only messages from steps s<=631 are ever popped

// LDS A-staging row strides (elements). Chosen so stride ≡ 8 dwords (mod 64):
// 16 rows tile the 64 LDS banks with the minimal 2-phase schedule for b128 loads.
#define STR2 272               // for 256-wide tiles (272*2B = 544B = 136 dwords, 136%64==8)
#define STR4 528               // for 512-wide tile  (528*2B = 1056B = 264 dwords, 264%64==8)

typedef __attribute__((ext_vector_type(16))) __bf16 v16bf;
typedef __attribute__((ext_vector_type(8)))  __bf16 v8bf;
typedef __attribute__((ext_vector_type(8)))  float  v8f;

static inline size_t h_alignup(size_t x) { return (x + 255) & ~(size_t)255; }

// K position within a 32-wide K-chunk for the bf16 16x16x32 fragment layout
__device__ __forceinline__ int frag_k(int lane, int e) {
    int v = e >> 1;
    return ((v >> 2) << 4) + ((lane >> 4) << 3) + ((v & 3) << 1) + (e & 1);
}

// A fragment = two contiguous 16B chunks: K = base+half*8 .. +7 and +16.. (+7).
// rowptr must point at stage[mrow*stride + col_off + kt*32] (16B aligned).
__device__ __forceinline__ v16bf load_fragA(const __bf16* rowptr, int half) {
    const v8bf lo = *(const v8bf*)(rowptr + half * 8);
    const v8bf hi = *(const v8bf*)(rowptr + 16 + half * 8);
    v16bf a;
    #pragma unroll
    for (int i = 0; i < 8; ++i) { a[i] = lo[i]; a[8 + i] = hi[i]; }
    return a;
}

// y[nt*16+lane] (valid on lanes 0..15, acc[0]) = x[0:K] @ W[:, nt*16+lane]
// rowbase = &stage[mrow*stride + col_off]; rows 1..15 of stage are zero.
__device__ __forceinline__ float wmma_matvec(const __bf16* __restrict__ sw,
                                             const __bf16* rowbase, int ktTiles,
                                             int nt, int lane, int half) {
    v8f acc = {0.f, 0.f, 0.f, 0.f, 0.f, 0.f, 0.f, 0.f};
    for (int kt = 0; kt < ktTiles; ++kt) {
        v16bf a = load_fragA(rowbase + kt * 32, half);
        v16bf b = *(const v16bf*)(sw + (size_t)((kt * 16 + nt) * 32 + lane) * 16);
        acc = __builtin_amdgcn_wmma_f32_16x16x32_bf16(
                  false, a, false, b, (short)0, acc, false, false);
    }
    return acc[0];
}

// ============================ K0: weight prep =================================
__global__ void gwac_prep(const float* __restrict__ Wr, const float* __restrict__ Wk,
                          const float* __restrict__ Wq, const float* __restrict__ Wm,
                          const float* __restrict__ Wd,
                          __bf16* __restrict__ swk, __bf16* __restrict__ swq,
                          __bf16* __restrict__ swr, __bf16* __restrict__ swm,
                          __bf16* __restrict__ swd) {
    int idx = blockIdx.x * 256 + threadIdx.x;       // 0..131071
    // Wm (K=512): e[0:3] lane[4:8] nt[9:12] kt[13:16]
    {
        int e = idx & 15, lane = (idx >> 4) & 31, nt = (idx >> 9) & 15, kt = idx >> 13;
        int k = kt * 32 + frag_k(lane, e);
        int n = nt * 16 + (lane & 15);
        swm[idx] = (__bf16)Wm[k * HH + n];
    }
    // K=256 weights: e[0:3] lane[4:8] nt[9:12] kt[13:15]
    if (idx < 65536) {
        int e = idx & 15, lane = (idx >> 4) & 31, nt = (idx >> 9) & 15, kt = idx >> 13;
        int k = kt * 32 + frag_k(lane, e);
        int n = nt * 16 + (lane & 15);
        swk[idx] = (__bf16)Wk[k * HH + n];
        swq[idx] = (__bf16)Wq[k * HH + n];
        // fold concat([values]*4) @ Wr into one 256x256 matrix during swizzle
        float r = Wr[k * HH + n] + Wr[(HH + k) * HH + n]
                + Wr[(2 * HH + k) * HH + n] + Wr[(3 * HH + k) * HH + n];
        swr[idx] = (__bf16)r;
    }
    // Wd: per p, e[0:3] lane[4:8] nt[9:10] kt[11:13]; p = bit 14
    if (idx < 32768) {
        int p = idx >> 14, w = idx & 16383;
        int e = w & 15, lane = (w >> 4) & 31, nt = (w >> 9) & 3, kt = (w >> 11) & 7;
        int k = kt * 32 + frag_k(lane, e);
        int n = nt * 16 + (lane & 15);
        swd[idx] = (__bf16)Wd[(p * HH + k) * OO + n];
    }
}

// ============================ K1: encode + hist init ==========================
__global__ void gwac_init(const float* __restrict__ xa, const float* __restrict__ We,
                          const float* __restrict__ be,
                          float* __restrict__ hist, __bf16* __restrict__ histb,
                          int* __restrict__ count) {
    int n = blockIdx.x, h = threadIdx.x;
    float acc = be[h];
    #pragma unroll 8
    for (int f = 0; f < FF; ++f) acc += xa[n * FF + f] * We[f * HH + h];
    hist[(n * HISTN + 0) * HH + h] = acc;
    histb[(n * HISTN + 0) * HH + h] = (__bf16)acc;
    for (int j = 1; j < HISTN; ++j) {
        hist[(n * HISTN + j) * HH + h] = 0.0f;
        histb[(n * HISTN + j) * HH + h] = (__bf16)0.0f;
    }
    if (h == 0) count[n] = 1;
}

// ============================ K2: node sequence ===============================
// pop at step t: t<64 -> node t ; else neighbors[node_{(t-64)>>3}][(t-64)&7]
__global__ void gwac_nodeseq(const int* __restrict__ neighbors, int* __restrict__ node_seq) {
    if (threadIdx.x == 0 && blockIdx.x == 0) {
        for (int t = 0; t < STEPS; ++t) {
            int nd = (t < SSTART) ? t
                   : neighbors[node_seq[(t - SSTART) >> 3] * DEGC + ((t - SSTART) & 7)];
            node_seq[t] = nd;
        }
    }
}

// ============================ K3: sequential walk =============================
__global__ void __launch_bounds__(256, 1)
gwac_walk(const float* __restrict__ first_message,
          const __bf16* __restrict__ swq, const float* __restrict__ bq,
          const __bf16* __restrict__ swk, const float* __restrict__ bk,
          const __bf16* __restrict__ swr, const float* __restrict__ br,
          const __bf16* __restrict__ swm, const float* __restrict__ bm,
          const int* __restrict__ node_seq,
          float* __restrict__ hist, __bf16* __restrict__ histb,
          int* __restrict__ count, float* __restrict__ msg_store) {
    // zero-padded A-staging tiles: row 0 = live vector, rows 1..15 stay zero
    __shared__ __align__(16) __bf16 s_featsb[16 * STR2];       // feats rows 0..9
    __shared__ __align__(16) __bf16 s_valA[16 * STR2];         // values vector
    __shared__ __align__(16) __bf16 s_catA[16 * STR4];         // [newstate | message]
    __shared__ float s_feats[HISTN][HH];
    __shared__ float s_keys[HISTN][HH];
    __shared__ float s_query[HH];
    __shared__ float s_logits[16];
    __shared__ float s_attn[16];

    const int tid  = threadIdx.x;
    const int wave = tid >> 5, lane = tid & 31;
    const int mrow = lane & 15, half = lane >> 4;
    const float inv_sqrt_h = 0.0625f;        // 1/sqrt(256)

    for (int i = tid; i < 16 * STR2; i += 256) { s_featsb[i] = (__bf16)0.0f; s_valA[i] = (__bf16)0.0f; }
    for (int i = tid; i < 16 * STR4; i += 256) s_catA[i] = (__bf16)0.0f;
    __syncthreads();

    const __bf16* fA  = s_featsb + mrow * STR2;   // keys A row base
    const __bf16* vA  = s_valA   + mrow * STR2;   // newstate A row base
    const __bf16* cA  = s_catA   + mrow * STR4;   // newmessage A row base
    const __bf16* qA  = cA + MMSZ;                // query A row base (message half)

    for (int t = 0; t < STEPS; ++t) {
        const int node = node_seq[t];
        const int c    = count[node];
        const int nv   = c < HISTN ? c : HISTN;
        const int slot = c % HISTN;
        // 8 consecutive pops share one message -> reload/query only on change
        const bool msg_new = (t < SSTART) || (((t - SSTART) & 7) == 0);

        if (msg_new) {
            float mv = (t < SSTART) ? first_message[t * MMSZ + tid]
                                    : msg_store[((t - SSTART) >> 3) * MMSZ + tid];
            s_catA[MMSZ + tid] = (__bf16)mv;      // row 0, message half
        }
        #pragma unroll
        for (int j = 0; j < HISTN; ++j) {
            float fv = hist[(node * HISTN + j) * HH + tid];
            s_feats[j][tid] = fv;
            s_featsb[j * STR2 + tid] = (__bf16)fv;
        }
        __syncthreads();                                       // (1)

        // query = message @ Wq + bq (WMMA mat-vec, amortized over 8 steps)
        if (msg_new) {
            #pragma unroll
            for (int nt = wave * 2; nt < wave * 2 + 2; ++nt) {
                float y = wmma_matvec(swq, qA, 8, nt, lane, half);
                if (lane < 16) {
                    int ncol = nt * 16 + lane;
                    s_query[ncol] = y + bq[ncol];
                }
            }
        }

        // keys = feats @ Wk + bk (16x256x256 WMMA, rows>=10 zero-padded)
        #pragma unroll
        for (int nt = wave * 2; nt < wave * 2 + 2; ++nt) {
            v8f acc = {0.f, 0.f, 0.f, 0.f, 0.f, 0.f, 0.f, 0.f};
            #pragma unroll
            for (int kt = 0; kt < 8; ++kt) {
                v16bf a = load_fragA(fA + kt * 32, half);
                v16bf b = *(const v16bf*)(swk + (size_t)((kt * 16 + nt) * 32 + lane) * 16);
                acc = __builtin_amdgcn_wmma_f32_16x16x32_bf16(
                          false, a, false, b, (short)0, acc, false, false);
            }
            int ncol = nt * 16 + (lane & 15);
            float bias = bk[ncol];
            #pragma unroll
            for (int r = 0; r < 8; ++r) {
                int m = r + (half << 3);
                if (m < HISTN) s_keys[m][ncol] = acc[r] + bias;
            }
        }
        __syncthreads();                                       // (2)

        // logits: one history slot per wave, lane-parallel dot + shuffle butterfly
        for (int j = wave; j < nv; j += 8) {
            float p = 0.f;
            #pragma unroll
            for (int h = lane; h < HH; h += 32) p += s_keys[j][h] * s_query[h];
            #pragma unroll
            for (int off = 16; off > 0; off >>= 1) p += __shfl_xor(p, off, 32);
            if (lane == 0) s_logits[j] = p * inv_sqrt_h;
        }
        __syncthreads();                                       // (3)
        if (tid == 0) {
            float mx = s_logits[0];
            for (int j = 1; j < nv; ++j) mx = fmaxf(mx, s_logits[j]);
            float ssum = 0.f;
            for (int j = 0; j < nv; ++j) { float e = __expf(s_logits[j] - mx); s_attn[j] = e; ssum += e; }
            float inv = 1.0f / ssum;
            for (int j = 0; j < nv; ++j) s_attn[j] *= inv;
        }
        __syncthreads();                                       // (4)

        // values = attn @ feats (f32), stage bf16 row 0 for newstate mat-vec
        {
            float v = 0.f;
            for (int j = 0; j < nv; ++j) v += s_attn[j] * s_feats[j][tid];
            s_valA[tid] = (__bf16)v;
        }
        __syncthreads();                                       // (5)

        // newstate = values @ Wr_sum + br (WMMA mat-vec) -> hist ring + catA
        #pragma unroll
        for (int nt = wave * 2; nt < wave * 2 + 2; ++nt) {
            float y = wmma_matvec(swr, vA, 8, nt, lane, half);
            if (lane < 16) {
                int ncol = nt * 16 + lane;
                float ns = y + br[ncol];
                hist[(node * HISTN + slot) * HH + ncol]  = ns;
                histb[(node * HISTN + slot) * HH + ncol] = (__bf16)ns;
                s_catA[ncol] = (__bf16)ns;        // row 0, newstate half
            }
        }
        __syncthreads();                                       // (6)

        // newmessage = concat(newstate, message) @ Wm + bm
        // only messages produced at t<640 are ever popped -> skip the rest
        if (t < MSG_SAVE) {
            #pragma unroll
            for (int nt = wave * 2; nt < wave * 2 + 2; ++nt) {
                float y = wmma_matvec(swm, cA, 16, nt, lane, half);
                if (lane < 16) {
                    int ncol = nt * 16 + lane;
                    msg_store[t * MMSZ + ncol] = y + bm[ncol];
                }
            }
        }
        if (tid == 0) count[node] = c + 1;
        __threadfence();
        __syncthreads();                                       // (7)
    }
}

// ============================ K4: gather final states =========================
__global__ void gwac_finalize(const float* __restrict__ hist, const int* __restrict__ count,
                              __bf16* __restrict__ finalb) {
    int n = blockIdx.x, h = threadIdx.x;
    int slot = (count[n] - 1) % HISTN;
    finalb[n * HH + h] = (__bf16)hist[(n * HISTN + slot) * HH + h];
}

// ============================ K5: decode + log_softmax ========================
__global__ void __launch_bounds__(128)
gwac_decode(const __bf16* __restrict__ finalb, const __bf16* __restrict__ swd,
            const float* __restrict__ bd, float* __restrict__ out) {
    __shared__ float s_log[16][OO];
    const int p  = blockIdx.x >> 5;
    const int rt = blockIdx.x & 31;                 // row tile: 16 rows of N
    const int tid = threadIdx.x;
    const int wave = tid >> 5, lane = tid & 31;     // 4 waves -> 4 N-tiles of 16
    const int mrow = rt * 16 + (lane & 15), half = lane >> 4;
    const __bf16* aBase = finalb + (size_t)mrow * HH;   // rows are 512B -> 16B aligned

    v8f acc = {0.f, 0.f, 0.f, 0.f, 0.f, 0.f, 0.f, 0.f};
    #pragma unroll
    for (int kt = 0; kt < 8; ++kt) {
        v16bf a = load_fragA(aBase + kt * 32, half);
        v16bf b = *(const v16bf*)(swd + (size_t)(((p * 8 + kt) * 4 + wave) * 32 + lane) * 16);
        acc = __builtin_amdgcn_wmma_f32_16x16x32_bf16(
                  false, a, false, b, (short)0, acc, false, false);
    }
    int ncol = wave * 16 + (lane & 15);
    float bias = bd[p * OO + ncol];
    #pragma unroll
    for (int r = 0; r < 8; ++r) s_log[r + (half << 3)][ncol] = acc[r] + bias;
    __syncthreads();

    if (tid < 16) {
        int row = tid;
        float mx = s_log[row][0];
        for (int o = 1; o < OO; ++o) mx = fmaxf(mx, s_log[row][o]);
        float ssum = 0.f;
        for (int o = 0; o < OO; ++o) ssum += __expf(s_log[row][o] - mx);
        float lse = mx + logf(ssum);
        int n = rt * 16 + row;
        for (int o = 0; o < OO; ++o)
            out[((size_t)p * NN + n) * OO + o] = s_log[row][o] - lse;
    }
}

// ============================ host launcher ===================================
extern "C" void kernel_launch(void* const* d_in, const int* in_sizes, int n_in,
                              void* d_out, int out_size, void* d_ws, size_t ws_size,
                              hipStream_t stream) {
    const float* xa      = (const float*)d_in[0];
    const int*   nbrs    = (const int*)  d_in[1];
    const float* firstm  = (const float*)d_in[2];
    const float* We      = (const float*)d_in[3];
    const float* be      = (const float*)d_in[4];
    const float* Wq      = (const float*)d_in[5];
    const float* bq      = (const float*)d_in[6];
    const float* Wk      = (const float*)d_in[7];
    const float* bk      = (const float*)d_in[8];
    const float* Wr      = (const float*)d_in[9];
    const float* br      = (const float*)d_in[10];
    const float* Wm      = (const float*)d_in[11];
    const float* bm      = (const float*)d_in[12];
    const float* Wd      = (const float*)d_in[13];
    const float* bd      = (const float*)d_in[14];
    (void)n_in; (void)in_sizes; (void)out_size; (void)ws_size;

    char* ws = (char*)d_ws;
    size_t off = 0;
    float*  hist      = (float*) (ws + off); off = h_alignup(off + (size_t)NN * HISTN * HH * 4);
    __bf16* histb     = (__bf16*)(ws + off); off = h_alignup(off + (size_t)NN * HISTN * HH * 2);
    int*    count     = (int*)   (ws + off); off = h_alignup(off + (size_t)NN * 4);
    int*    node_seq  = (int*)   (ws + off); off = h_alignup(off + (size_t)STEPS * 4);
    float*  msg_store = (float*) (ws + off); off = h_alignup(off + (size_t)MSG_SAVE * MMSZ * 4);
    __bf16* swk       = (__bf16*)(ws + off); off = h_alignup(off + (size_t)65536  * 2);
    __bf16* swq       = (__bf16*)(ws + off); off = h_alignup(off + (size_t)65536  * 2);
    __bf16* swr       = (__bf16*)(ws + off); off = h_alignup(off + (size_t)65536  * 2);
    __bf16* swm       = (__bf16*)(ws + off); off = h_alignup(off + (size_t)131072 * 2);
    __bf16* swd       = (__bf16*)(ws + off); off = h_alignup(off + (size_t)32768  * 2);
    __bf16* finalb    = (__bf16*)(ws + off); off = h_alignup(off + (size_t)NN * HH * 2);

    float* out = (float*)d_out;

    gwac_prep    <<<512, 256, 0, stream>>>(Wr, Wk, Wq, Wm, Wd, swk, swq, swr, swm, swd);
    gwac_init    <<<NN,  256, 0, stream>>>(xa, We, be, hist, histb, count);
    gwac_nodeseq <<<1,   1,   0, stream>>>(nbrs, node_seq);
    gwac_walk    <<<1,   256, 0, stream>>>(firstm, swq, bq, swk, bk, swr, br,
                                           swm, bm, node_seq, hist, histb, count, msg_store);
    gwac_finalize<<<NN,  256, 0, stream>>>(hist, count, finalb);
    gwac_decode  <<<PPSZ * 32, 128, 0, stream>>>(finalb, swd, bd, out);
}
